// LongConv_7146825581153
// MI455X (gfx1250) — matrix operand order repaired and checked by reading, not compiled, not run on hardware
//
#include <hip/hip_runtime.h>

typedef __attribute__((ext_vector_type(16))) __bf16 v16bf;
typedef __attribute__((ext_vector_type(8)))  float  v8f;

union Frag16 {
    v16bf          v;
    unsigned short u16[16];
    unsigned int   u32[8];
    int4           q[2];
};

__device__ __forceinline__ unsigned short f2bf(float f) {
    unsigned int x = __float_as_uint(f);
    x += 0x7FFFu + ((x >> 16) & 1u);           // round-to-nearest-even
    return (unsigned short)(x >> 16);
}

__device__ __forceinline__ float softthresh(float x) {
    float m = fabsf(x) - 0.1f;
    m = m > 0.f ? m : 0.f;
    return copysignf(m, x);
}

// problem dims
#define BB 4
#define HH 512
#define LL 4096
#define CC 2
#define L2 8192
#define L2M 8191
#define FF 1024
#define GG 1024

// stage-1 tiling
#define UPAD   112     // covers max left shift 16*7
#define UBUFN  4352    // 112 + 4192+8+16+31 rounded, 16B-aligned rows
#define KSTEPS 132     // m_base up to 4192 covers m < 4096+16*7

// ---------------------------------------------------------------------------
// Stage 0: W_out f32 -> bf16
// ---------------------------------------------------------------------------
__global__ __launch_bounds__(256) void prep_w(const float* __restrict__ W,
                                              unsigned short* __restrict__ Wbf) {
    int i = blockIdx.x * 256 + threadIdx.x;
    if (i < GG * FF) Wbf[i] = f2bf(W[i]);
}

// ---------------------------------------------------------------------------
// Stage 1: Toeplitz-GEMM long conv (bf16 WMMA, f32 acc) + skip + GELU
// grid = C*H*2 blocks (each block: one (c,h) row, 2 of the 4 batch entries)
// A rows = (b_local, shift r in 0..7); B = Toeplitz tile of reversed kernel;
// D cols = 16 consecutive n; tile covers 128 outputs with zero redundancy.
// ---------------------------------------------------------------------------
__global__ __launch_bounds__(256) void conv_wmma(
    const float* __restrict__ u, const float* __restrict__ kin,
    const float* __restrict__ Dp, unsigned short* __restrict__ ybf)
{
    __shared__ unsigned int kev[4096];                       // (kr[2i],   kr[2i+1]) packed
    __shared__ unsigned int kod[4096];                       // (kr[2i+1], kr[2i+2]) packed
    __shared__ __align__(16) unsigned short ubuf[2][UBUFN];  // zero-padded bf16 u

    const int tid = threadIdx.x;
    const int bid = blockIdx.x;
    const int ch  = bid >> 1;          // feature f = c*H + h
    const int bg  = bid & 1;           // batch pair {2bg, 2bg+1}
    const int h   = ch & (HH - 1);

    // reversed, soft-thresholded kernel row; dual-parity u32 packing so any
    // consecutive bf16 pair is one aligned ds_load_b32
    const float* krow = kin + (size_t)ch * L2;
    for (int i = tid; i < 4096; i += 256) {
        int t0 = 2 * i;
        float a0 = krow[(L2 - t0)     & L2M];
        float a1 = krow[(L2 - t0 - 1) & L2M];
        float a2 = krow[(L2 - t0 - 2) & L2M];
        unsigned short b0 = f2bf(softthresh(a0));
        unsigned short b1 = f2bf(softthresh(a1));
        unsigned short b2 = f2bf(softthresh(a2));
        kev[i] = (unsigned int)b0 | ((unsigned int)b1 << 16);
        kod[i] = (unsigned int)b1 | ((unsigned int)b2 << 16);
    }
    // zero-padded bf16 copy of the two u rows
    for (int i = tid; i < 2 * UBUFN; i += 256) {
        int bl = i / UBUFN;
        int mm = i - bl * UBUFN;
        int m  = mm - UPAD;
        int b  = 2 * bg + bl;
        float v = (m >= 0 && m < LL) ? u[(size_t)(b * HH + h) * LL + m] : 0.f;
        ubuf[bl][mm] = f2bf(v);
    }
    __syncthreads();

    const int wid  = tid >> 5;
    const int lane = tid & 31;
    const int mrow = lane & 15;        // A/D row index M, also D column via C/D layout
    const int hi   = lane >> 4;        // K half select
    const int ab   = mrow >> 3;        // A row -> local batch
    const int ar   = mrow & 7;         // A row -> shift r
    const float Dch = Dp[ch];

    // Toeplitz word parity is (mrow & 1): hoist the parity-array select
    const unsigned int* kbase = (mrow & 1) ? kod : kev;

    v8f zero = {0.f,0.f,0.f,0.f,0.f,0.f,0.f,0.f};
    v8f acc[4] = {zero, zero, zero, zero};

    for (int ks = 0; ks < KSTEPS; ++ks) {
        const int mb = ks * 32;
        // A fragment: shifted u, aligned b128 loads (shared by all 4 tiles)
        Frag16 a;
        const int e0 = UPAD + mb - 16 * ar + 16 * hi;  // multiple of 8 -> 16B aligned
        a.q[0] = *(const int4*)&ubuf[ab][e0];          // K {0..7}+16hi
        a.q[1] = *(const int4*)&ubuf[ab][e0 + 8];      // K {8..15}+16hi

        #pragma unroll
        for (int ti = 0; ti < 4; ++ti) {
            const int na = 128 * (wid * 4 + ti) + UPAD;
            const int wb = (mb + 16 * hi - na - mrow) >> 1;  // u32-word base (mod 4096)
            Frag16 bb;
            #pragma unroll
            for (int v = 0; v < 8; ++v) {
                bb.u32[v] = kbase[(wb + v) & 4095];
            }
            acc[ti] = __builtin_amdgcn_wmma_f32_16x16x32_bf16(
                false, a.v, false, bb.v, (short)0, acc[ti], false, false);
        }
    }

    // epilogue: skip (f32) + exact GELU + bf16 store in [f][(b,l)] layout
    #pragma unroll
    for (int ti = 0; ti < 4; ++ti) {
        const int na = 128 * (wid * 4 + ti) + UPAD;
        #pragma unroll
        for (int qq = 0; qq < 8; ++qq) {
            int row = qq + 8 * hi;                     // D: M = q + 8*hi
            int b   = 2 * bg + (row >> 3);
            int r   = row & 7;
            int n   = na + mrow - 16 * r;              // in [128t, 128t+128)
            float val = acc[ti][qq] + u[(size_t)(b * HH + h) * LL + n] * Dch;
            float ge  = 0.5f * val * (1.f + erff(val * 0.70710678118f));
            ybf[(size_t)ch * (BB * LL) + b * LL + n] = f2bf(ge);
        }
    }
}

// ---------------------------------------------------------------------------
// Stage 2: z[(b,l), g] = GELU(y)[(b,l), f] @ W[g, f]^T + bias  (bf16 WMMA)
// block = 128 rows x 64 cols, 8 waves (16 rows each, 4 col-tiles each)
// A tile is DMA'd into LDS with CDNA5 async global->LDS copies (ASYNCcnt).
// ---------------------------------------------------------------------------
__global__ __launch_bounds__(256) void gemm_out(
    const unsigned short* __restrict__ ybf, const unsigned short* __restrict__ Wbf,
    const float* __restrict__ bout, float* __restrict__ zbuf)
{
    __shared__ unsigned short asmem[32][128];   // [f within chunk][row within block]
    const int tid  = threadIdx.x;
    const int g0   = blockIdx.x * 64;
    const int row0 = blockIdx.y * 128;
    const int wid  = tid >> 5;
    const int lane = tid & 31;
    const int mrow = lane & 15;
    const int hi   = lane >> 4;

    // raw LDS offset = low 32 bits of the generic shared address (ISA 10.2)
    const unsigned asbase = (unsigned)(uintptr_t)(&asmem[0][0]);
    const char*    ysrc   = (const char*)ybf;

    v8f zero = {0.f,0.f,0.f,0.f,0.f,0.f,0.f,0.f};
    v8f acc[4] = {zero, zero, zero, zero};

    for (int kb = 0; kb < 32; ++kb) {
        const int f0 = kb * 32;
        __syncthreads();                               // protect previous tile reads
        // async DMA: 2048 dwords, one b32 per lane per step, no VGPR round-trip
        #pragma unroll
        for (int it = 0; it < 8; ++it) {
            int idx = tid + 256 * it;                  // dword index in 32x128 tile
            int f   = idx >> 6;                        // 64 dwords per f-row
            int ww  = idx & 63;
            unsigned voff  = (unsigned)(f0 + f) * (BB * LL * 2) +
                             (unsigned)row0 * 2 + (unsigned)ww * 4;
            unsigned laddr = asbase + (unsigned)idx * 4;
            asm volatile("global_load_async_to_lds_b32 %0, %1, %2"
                         :: "v"(laddr), "v"(voff), "s"(ysrc) : "memory");
        }
        asm volatile("s_wait_asynccnt 0x0" ::: "memory");
        __syncthreads();

        Frag16 a;                                      // transposed read from LDS
        #pragma unroll
        for (int i = 0; i < 16; ++i) {
            int kk = i + 16 * hi;                      // A layout: {0..15}+16hi
            a.u16[i] = asmem[kk][wid * 16 + mrow];
        }
        #pragma unroll
        for (int ct = 0; ct < 4; ++ct) {
            int g = g0 + ct * 16 + mrow;
            Frag16 bb;                                 // contiguous, aligned b128 pair
            bb.q[0] = *(const int4*)&Wbf[(size_t)g * FF + f0 + 16 * hi];
            bb.q[1] = *(const int4*)&Wbf[(size_t)g * FF + f0 + 16 * hi + 8];
            acc[ct] = __builtin_amdgcn_wmma_f32_16x16x32_bf16(
                false, a.v, false, bb.v, (short)0, acc[ct], false, false);
        }
    }

    #pragma unroll
    for (int ct = 0; ct < 4; ++ct) {
        int g = g0 + ct * 16 + mrow;
        float bias = bout[g];
        #pragma unroll
        for (int qq = 0; qq < 8; ++qq) {
            int row = row0 + wid * 16 + qq + 8 * hi;
            zbuf[(size_t)row * GG + g] = acc[ct][qq] + bias;
        }
    }
}

// ---------------------------------------------------------------------------
// Stage 3: GLU + transpose [B,L,H] -> [B,H,L] via 32x32 LDS tiles
// ---------------------------------------------------------------------------
__global__ __launch_bounds__(256) void glu_transpose(
    const float* __restrict__ zbuf, float* __restrict__ out)
{
    __shared__ float tile[32][33];
    const int h0 = blockIdx.x * 32;
    const int l0 = blockIdx.y * 32;
    const int b  = blockIdx.z;
    const int tid = threadIdx.x;

    #pragma unroll
    for (int e = 0; e < 4; ++e) {
        int i  = tid + 256 * e;
        int li = i >> 5;
        int hh = i & 31;
        const float* zr = zbuf + (size_t)(b * LL + l0 + li) * GG + h0 + hh;
        float av = zr[0];
        float gv = zr[HH];
        tile[li][hh] = av * (1.f / (1.f + expf(-gv)));
    }
    __syncthreads();
    #pragma unroll
    for (int e = 0; e < 4; ++e) {
        int i  = tid + 256 * e;
        int hh = i >> 5;
        int li = i & 31;
        out[(size_t)(b * HH + h0 + hh) * LL + l0 + li] = tile[li][hh];
    }
}

// ---------------------------------------------------------------------------
extern "C" void kernel_launch(void* const* d_in, const int* in_sizes, int n_in,
                              void* d_out, int out_size, void* d_ws, size_t ws_size,
                              hipStream_t stream) {
    (void)in_sizes; (void)n_in; (void)out_size; (void)ws_size;
    const float* u    = (const float*)d_in[0];   // [B,H,L]
    const float* kin  = (const float*)d_in[1];   // [C,H,2L]
    const float* Dp   = (const float*)d_in[2];   // [C,H]
    const float* Wout = (const float*)d_in[3];   // [2H, C*H]
    const float* bout = (const float*)d_in[4];   // [2H]
    float* out = (float*)d_out;                  // [B,H,L]

    char* wsb = (char*)d_ws;
    unsigned short* ybf = (unsigned short*)wsb;                         // 32 MB  [F][B*L]
    unsigned short* Wbf = (unsigned short*)(wsb + ((size_t)32 << 20));  //  2 MB  [G][F]
    float*          zb  = (float*)(wsb + ((size_t)34 << 20));           // 64 MB  [B*L][G]

    prep_w<<<(GG * FF + 255) / 256, 256, 0, stream>>>(Wout, Wbf);
    conv_wmma<<<CC * HH * 2, 256, 0, stream>>>(u, kin, Dp, ybf);
    gemm_out<<<dim3(GG / 64, (BB * LL) / 128), 256, 0, stream>>>(ybf, Wbf, bout, zb);
    glu_transpose<<<dim3(HH / 32, LL / 32, BB), 256, 0, stream>>>(zb, out);
}
